// StructuralMambaMixer_81698867905020
// MI455X (gfx1250) — matrix operand order repaired and checked by reading, not compiled, not run on hardware
//
#include <hip/hip_runtime.h>
#include <hip/hip_bf16.h>

// ---- problem constants (from reference) ----
#define DM     1024      // d_model
#define DI     2048      // d_inner
#define DSTATE 16
#define DTRANK 64
#define BB     2
#define LL     2048
#define MROWS  (BB*LL)   // 4096 rows in every GEMM
#define XDBL_W (DTRANK + 2*DSTATE)   // 96

typedef _Float16     v16h __attribute__((ext_vector_type(16)));
typedef float        v8f  __attribute__((ext_vector_type(8)));
typedef unsigned int u4   __attribute__((ext_vector_type(4)));

union Frag { struct { u4 lo, hi; } q; v16h v; };   // one WMMA A/B operand

// ---- CDNA5 async global->LDS copy (ASYNCcnt) -------------------------------
__device__ __forceinline__ void async_copy_b128(unsigned lds_addr, const void* gptr) {
    unsigned long long ga = (unsigned long long)(size_t)gptr;
    asm volatile("global_load_async_to_lds_b128 %0, %1, off"
                 :: "v"(lds_addr), "v"(ga) : "memory");
}
__device__ __forceinline__ void wait_async_le(int n) {
    if (n == 0) asm volatile("s_wait_asynccnt 0x0" ::: "memory");
    else        asm volatile("s_wait_asynccnt 0x3" ::: "memory");
}

// ----------------------------------------------------------------------------
// All-f16 WMMA GEMM with double-buffered async LDS tiles:
//   C[M,N](f32) = A[M,K(lda)](f16) * W[N,K(ldb)](f16)^T
// WG = 256 threads (8 waves), WG tile 64(M) x 128(N), K step 32.
// EPI: 0 = plain store, 1 = softplus(acc + bias[n])
// ----------------------------------------------------------------------------
template<int EPI>
__global__ __launch_bounds__(256)
void wmma_gemm_h(const _Float16* __restrict__ A, int lda,
                 const _Float16* __restrict__ W, int ldb,
                 float* __restrict__ C, int ldc,
                 int N, int K, const float* __restrict__ bias)
{
    // per buffer: A 64x32 f16 (4KB) + B 128x32 f16 (8KB) = 12KB; double buffered
    __shared__ __align__(16) _Float16 lds[2 * 6144];

    const int tid    = threadIdx.x;
    const int lane   = tid & 31;
    const int wave   = tid >> 5;
    const int waveM  = wave >> 2;   // 0..1
    const int waveN  = wave & 3;    // 0..3
    const int lane16 = lane & 15;
    const int half   = lane >> 4;
    const int rowBase = blockIdx.y * 64;
    const int colBase = blockIdx.x * 128;

    // visible zero-init of both buffers (defined data everywhere)
    {
        u4* z = (u4*)lds;
        #pragma unroll
        for (int i = 0; i < 6; ++i) {
            u4 zz; zz[0] = 0u; zz[1] = 0u; zz[2] = 0u; zz[3] = 0u;
            z[tid + 256 * i] = zz;
        }
    }
    __syncthreads();

    const unsigned ldsBase = (unsigned)(size_t)&lds[0];   // LDS byte address

    // cooperative loader geometry (3 x b128 async per thread per K-tile)
    const int aRow = tid >> 2, aCh = tid & 3;   // A: 64 rows x 4 chunks x 16B
    const int bRow = tid >> 1, bCh = tid & 1;   // B: 128 rows x 2 chunks x 32B
    const int bN   = colBase + bRow;
    const int bNc  = (bN < N) ? bN : (N - 1);   // clamp: no EXEC divergence; masked at store

    const _Float16* aSrc = A + (size_t)(rowBase + aRow) * lda + aCh * 8;
    const _Float16* bSrc = W + (size_t)bNc * ldb + bCh * 16;
    const unsigned aDst = (unsigned)(aRow * 64 + aCh * 16);          // bytes (A region)
    const unsigned bDst = (unsigned)(4096 + bRow * 64 + bCh * 32);   // bytes (B region)

    const int kTiles = K >> 5;

    // prologue: tile 0 -> buffer 0
    async_copy_b128(ldsBase + aDst, aSrc);
    async_copy_b128(ldsBase + bDst,      bSrc);
    async_copy_b128(ldsBase + bDst + 16, bSrc + 8);

    v8f acc[2][2] = {};

    for (int kt = 0; kt < kTiles; ++kt) {
        if (kt + 1 < kTiles) {  // issue next tile into the other buffer
            const unsigned bb = ldsBase + ((unsigned)((kt + 1) & 1)) * 12288u;
            const int kOff = (kt + 1) << 5;   // halfs
            async_copy_b128(bb + aDst, aSrc + kOff);
            async_copy_b128(bb + bDst,      bSrc + kOff);
            async_copy_b128(bb + bDst + 16, bSrc + kOff + 8);
            wait_async_le(3);   // in-order => tile kt's 3 copies complete
        } else {
            wait_async_le(0);
        }
        __syncthreads();        // tile kt resident WG-wide

        const unsigned* bufU = (const unsigned*)&lds[(kt & 1) * 6144];
        Frag aF[2], bF[2];
        #pragma unroll
        for (int m = 0; m < 2; ++m) {          // A frag: lane=M, VGPR v -> K=(v/4)*16+half*8+(v%4)*2
            const int r = waveM * 32 + m * 16 + lane16;
            const unsigned* p = bufU + r * 16 + half * 4;
            aF[m].q.lo = *(const u4*)p;
            aF[m].q.hi = *(const u4*)(p + 8);
        }
        #pragma unroll
        for (int n = 0; n < 2; ++n) {          // B frag: lane=N, VGPR v -> K=(v/4)*16+half*8+(v%4)*2
            const int c = waveN * 32 + n * 16 + lane16;
            const unsigned* p = bufU + 1024 + c * 16 + half * 4;
            bF[n].q.lo = *(const u4*)p;
            bF[n].q.hi = *(const u4*)(p + 8);
        }
        #pragma unroll
        for (int m = 0; m < 2; ++m)
            #pragma unroll
            for (int n = 0; n < 2; ++n)
                acc[m][n] = __builtin_amdgcn_wmma_f32_16x16x32_f16(
                    false, aF[m].v, false, bF[n].v, (short)0, acc[m][n], false, false);

        __syncthreads();        // all waves done reading buf[kt&1] before it is reused
    }

    // epilogue: C/D layout lane=N, VGPR r -> M = r + 8*half
    #pragma unroll
    for (int m = 0; m < 2; ++m) {
        const int rBase = rowBase + waveM * 32 + m * 16 + half * 8;
        #pragma unroll
        for (int n = 0; n < 2; ++n) {
            const int col = colBase + waveN * 32 + n * 16 + lane16;
            if (col < N) {
                const float bv = (EPI == 1) ? bias[col] : 0.f;
                #pragma unroll
                for (int r = 0; r < 8; ++r) {
                    float v = acc[m][n][r];
                    if (EPI == 1) {
                        v += bv;
                        v = (v > 15.f) ? v : __logf(1.f + __expf(v));  // softplus
                    }
                    C[(size_t)(rBase + r) * ldc + col] = v;
                }
            }
        }
    }
}

// ----------------------------------------------------------------------------
// fp32 -> fp16 conversion pre-pass
// ----------------------------------------------------------------------------
__global__ __launch_bounds__(256)
void cvt_f32_f16(const float* __restrict__ s, _Float16* __restrict__ d, int n)
{
    const int i = blockIdx.x * 256 + threadIdx.x;
    if (i < n) d[i] = (_Float16)s[i];
}

// ----------------------------------------------------------------------------
// Depthwise causal conv1d (k=4) + SiLU. x lives in proj[:, 0:DI] (row stride 2*DI).
// Writes u (f32, for the scan) and u_h (f16, A operand of x_proj GEMM).
// ----------------------------------------------------------------------------
__global__ __launch_bounds__(256)
void conv_silu(const float* __restrict__ xin, const float* __restrict__ w,
               const float* __restrict__ b, float* __restrict__ u,
               _Float16* __restrict__ u_h)
{
    const int idx = blockIdx.x * 256 + threadIdx.x;
    const int row = idx / DI;            // b*L + l
    const int d   = idx - row * DI;
    const int l   = row & (LL - 1);
    float acc = b[d];
    const float* wr = w + d * 4;
    #pragma unroll
    for (int j = 0; j < 4; ++j) {
        const int ll = l - 3 + j;
        if (ll >= 0)
            acc = fmaf(xin[(size_t)(row - 3 + j) * (2 * DI) + d], wr[j], acc);
    }
    const float s = acc / (1.f + __expf(-acc));   // SiLU
    u[(size_t)row * DI + d] = s;
    u_h[(size_t)row * DI + d] = (_Float16)s;
}

// ----------------------------------------------------------------------------
// Selective scan: one thread per (batch, channel); h[16] in registers.
// B_t/C_t broadcast via LDS. Writes y*silu(gate) as f16 (out_proj A operand).
// ----------------------------------------------------------------------------
__global__ __launch_bounds__(256)
void selective_scan(const float* __restrict__ dt, const float* __restrict__ u,
                    const float* __restrict__ xdbl, const float* __restrict__ proj,
                    const float* __restrict__ A_log, const float* __restrict__ D_skip,
                    _Float16* __restrict__ y_h)
{
    const int b = blockIdx.x >> 3;                     // DI/256 = 8 blocks per batch
    const int d = (blockIdx.x & 7) * 256 + threadIdx.x;
    __shared__ float sB[DSTATE], sC[DSTATE];

    float Arow[DSTATE];
    #pragma unroll
    for (int s = 0; s < DSTATE; ++s) Arow[s] = -__expf(A_log[d * DSTATE + s]);
    const float Dsk = D_skip[d];
    float h[DSTATE];
    #pragma unroll
    for (int s = 0; s < DSTATE; ++s) h[s] = 0.f;

    for (int l = 0; l < LL; ++l) {
        const int row = b * LL + l;
        __syncthreads();
        if (threadIdx.x < 2 * DSTATE) {
            const float v = xdbl[row * XDBL_W + DTRANK + threadIdx.x];
            if (threadIdx.x < DSTATE) sB[threadIdx.x] = v;
            else                      sC[threadIdx.x - DSTATE] = v;
        }
        __syncthreads();
        const float dtv = dt[(size_t)row * DI + d];
        const float uv  = u [(size_t)row * DI + d];
        const float du  = dtv * uv;
        float y = 0.f;
        #pragma unroll
        for (int s = 0; s < DSTATE; ++s) {
            h[s] = __expf(dtv * Arow[s]) * h[s] + sB[s] * du;   // dA*h + dt*B*u
            y = fmaf(h[s], sC[s], y);
        }
        const float g = proj[(size_t)row * (2 * DI) + DI + d];  // gate
        y_h[(size_t)row * DI + d] =
            (_Float16)((y + uv * Dsk) * (g / (1.f + __expf(-g))));
    }
}

// ----------------------------------------------------------------------------
extern "C" void kernel_launch(void* const* d_in, const int* in_sizes, int n_in,
                              void* d_out, int out_size, void* d_ws, size_t ws_size,
                              hipStream_t stream)
{
    const float* hidden     = (const float*)d_in[0];
    // d_in[1] = time_deltas (unused by reference math)
    const float* in_proj_w  = (const float*)d_in[2];
    const float* conv_w     = (const float*)d_in[3];
    const float* conv_b     = (const float*)d_in[4];
    const float* x_proj_w   = (const float*)d_in[5];
    const float* dt_proj_w  = (const float*)d_in[6];
    const float* dt_proj_b  = (const float*)d_in[7];
    const float* A_log      = (const float*)d_in[8];
    const float* D_skip     = (const float*)d_in[9];
    const float* out_proj_w = (const float*)d_in[10];
    float* out = (float*)d_out;

    // ---- workspace layout ----
    float* proj = (float*)d_ws;                          // [4096][4096]
    float* u    = proj + (size_t)MROWS * (2 * DI);       // [4096][2048]
    float* xdbl = u    + (size_t)MROWS * DI;             // [4096][96]
    float* dt   = xdbl + (size_t)MROWS * XDBL_W;         // [4096][2048]
    _Float16* hidden_h = (_Float16*)(dt + (size_t)MROWS * DI);
    _Float16* inW_h    = hidden_h + (size_t)MROWS * DM;      // [4096][1024]
    _Float16* xW_h     = inW_h    + (size_t)(2 * DI) * DM;   // [96][2048]
    _Float16* dtW_h    = xW_h     + (size_t)XDBL_W * DI;     // [2048][64]
    _Float16* outW_h   = dtW_h    + (size_t)DI * DTRANK;     // [1024][2048]
    _Float16* u_h      = outW_h   + (size_t)DM * DI;         // [4096][2048]
    _Float16* xdbl_h   = u_h      + (size_t)MROWS * DI;      // [4096][96]
    _Float16* y_h      = xdbl_h   + (size_t)MROWS * XDBL_W;  // [4096][2048]

    const dim3 blk(256);
    auto cvt = [&](const float* s, _Float16* d, int n) {
        cvt_f32_f16<<<(n + 255) / 256, blk, 0, stream>>>(s, d, n);
    };

    // 0) one-time f16 conversions (activations input + all GEMM weights)
    cvt(hidden,     hidden_h, MROWS * DM);
    cvt(in_proj_w,  inW_h,    2 * DI * DM);
    cvt(x_proj_w,   xW_h,     XDBL_W * DI);
    cvt(dt_proj_w,  dtW_h,    DI * DTRANK);
    cvt(out_proj_w, outW_h,   DM * DI);

    // 1) in_proj: proj[M,4096] = hidden @ in_proj_w^T
    wmma_gemm_h<0><<<dim3((2 * DI) / 128, MROWS / 64), blk, 0, stream>>>(
        hidden_h, DM, inW_h, DM, proj, 2 * DI, 2 * DI, DM, nullptr);

    // 2) causal depthwise conv + SiLU -> u (f32) and u_h (f16)
    conv_silu<<<(MROWS * DI) / 256, blk, 0, stream>>>(proj, conv_w, conv_b, u, u_h);

    // 3) x_proj: xdbl[M,96] = u @ x_proj_w^T
    wmma_gemm_h<0><<<dim3(1, MROWS / 64), blk, 0, stream>>>(
        u_h, DI, xW_h, DI, xdbl, XDBL_W, XDBL_W, DI, nullptr);
    cvt(xdbl, xdbl_h, MROWS * XDBL_W);

    // 4) dt = softplus(xdbl[:, :64] @ dt_proj_w^T + dt_proj_b)
    wmma_gemm_h<1><<<dim3(DI / 128, MROWS / 64), blk, 0, stream>>>(
        xdbl_h, XDBL_W, dtW_h, DTRANK, dt, DI, DI, DTRANK, dt_proj_b);

    // 5) selective scan -> y_h (f16), gated by silu(gate) from proj[:, DI:]
    selective_scan<<<BB * (DI / 256), blk, 0, stream>>>(
        dt, u, xdbl, proj, A_log, D_skip, y_h);

    // 6) out_proj: out[M,1024] = y @ out_proj_w^T
    wmma_gemm_h<0><<<dim3(DM / 128, MROWS / 64), blk, 0, stream>>>(
        y_h, DI, outW_h, DI, out, DM, DM, DI, nullptr);
}
